// DeViSE_53541062312223
// MI455X (gfx1250) — compile-verified, matrix operand order checked by reading, not run on hardware
//
#include <hip/hip_runtime.h>
#include <hip/hip_bf16.h>

// ---------------------------------------------------------------------------
// DeViSE forward on MI455X (gfx1250): 4 GEMMs via V_WMMA_F32_16X16X32_BF16,
// LDS-tiled using CDNA5 async global->LDS copies (ASYNCcnt) via inline asm.
// fp32 inputs converted once to bf16 (RNE); accumulation stays fp32.
// ---------------------------------------------------------------------------

typedef __bf16 bf16_t;
typedef __attribute__((ext_vector_type(8)))  __bf16 v8bf;
typedef __attribute__((ext_vector_type(16))) __bf16 v16bf;
typedef __attribute__((ext_vector_type(8)))  float  v8f;

// Async 16-byte global -> LDS copy (GLOBAL_LOAD_ASYNC_TO_LDS_B128, GV mode).
// VDST = 32-bit LDS byte offset (generic LDS ptr truncates to addr[31:0]),
// VADDR = 64-bit global address, SADDR = off.  Tracked by ASYNCcnt.
__device__ __forceinline__ void cp16_g2l(const bf16_t* g, bf16_t* l) {
    unsigned           lofs = (unsigned)(size_t)l;
    unsigned long long gaddr = (unsigned long long)g;
    asm volatile("global_load_async_to_lds_b128 %0, %1, off"
                 :: "v"(lofs), "v"(gaddr) : "memory");
}

__device__ __forceinline__ void wait_async_lds() {
    asm volatile("s_wait_asynccnt 0x0" ::: "memory");
}

__device__ __forceinline__ bf16_t cvt_bf16(float f) {
    union { float f; unsigned u; } x; x.f = f;
    unsigned u = x.u;
    u += 0x7FFFu + ((u >> 16) & 1u);      // round-to-nearest-even
    unsigned short hs = (unsigned short)(u >> 16);
    bf16_t r;
    __builtin_memcpy(&r, &hs, 2);
    return r;
}

// fp32 [srows x scols] -> bf16 [drows x dcols], zero padded.
__global__ __launch_bounds__(256) void cvt_pad_bf16(
    const float* __restrict__ src, int srows, int scols,
    bf16_t* __restrict__ dst, int drows, int dcols)
{
    int idx = blockIdx.x * blockDim.x + threadIdx.x;
    int total = drows * dcols;
    if (idx >= total) return;
    int r = idx / dcols;
    int c = idx - r * dcols;
    float v = (r < srows && c < scols) ? src[(size_t)r * scols + c] : 0.0f;
    dst[idx] = cvt_bf16(v);
}

// C = act(A @ B^T + bias).  A: [M x K] bf16 row-major, B: [Npad x K] bf16
// row-major (transposed operand).  M % 128 == 0, Npad % 128 == 0, K % 32 == 0.
// Block = 256 threads = 8 waves (4 in M x 2 in N); block tile 128x128,
// wave tile 32x64, BK = 32, double-buffered LDS.
#define LDS_STRIDE 40   // 32 bf16 payload + 8 pad -> conflict-free ds_load_b128

__global__ __launch_bounds__(256) void gemm_bf16_wmma(
    const bf16_t* __restrict__ A,
    const bf16_t* __restrict__ B,
    const float*  __restrict__ bias,   // [>= Nreal] or nullptr
    float*        __restrict__ Cf,     // [M x Nreal] fp32 out, or nullptr
    bf16_t*       __restrict__ Cb,     // [M x Npad]  bf16 out, or nullptr
    int M, int Npad, int Nreal, int K, int relu)
{
    __shared__ __align__(16) bf16_t shA[2][128 * LDS_STRIDE];
    __shared__ __align__(16) bf16_t shB[2][128 * LDS_STRIDE];

    const int nBN  = Npad >> 7;
    const int blkM = blockIdx.x / nBN;
    const int blkN = blockIdx.x - blkM * nBN;

    const int t     = threadIdx.x;
    const int wid   = t >> 5;            // 0..7
    const int waveM = wid >> 1;          // 0..3 -> 32-row slice
    const int waveN = wid & 1;           // 0..1 -> 64-col slice
    const int lane  = t & 31;
    const int h     = lane >> 4;         // K half per ISA 7.12.2 16-bit layout
    const int l15   = lane & 15;

    // ---- staging: each thread moves 4x16B per K-step (2 for A, 2 for B) ----
    const int sRow = t >> 2;             // 0..63
    const int sOff = (t & 3) * 8;        // 0,8,16,24 elements within BK=32
    const bf16_t* gA0 = A + (size_t)(blkM * 128 + sRow) * K + sOff;
    const bf16_t* gA1 = gA0 + (size_t)64 * K;
    const bf16_t* gB0 = B + (size_t)(blkN * 128 + sRow) * K + sOff;
    const bf16_t* gB1 = gB0 + (size_t)64 * K;
    const int lOff0 = sRow * LDS_STRIDE + sOff;
    const int lOff1 = (sRow + 64) * LDS_STRIDE + sOff;

    v8f acc[2][4];
    const v8f vzero = {};
#pragma unroll
    for (int mi = 0; mi < 2; ++mi)
#pragma unroll
        for (int ni = 0; ni < 4; ++ni)
            acc[mi][ni] = vzero;

    // prologue: fill buffer 0
    cp16_g2l(gA0, &shA[0][lOff0]);
    cp16_g2l(gA1, &shA[0][lOff1]);
    cp16_g2l(gB0, &shB[0][lOff0]);
    cp16_g2l(gB1, &shB[0][lOff1]);
    wait_async_lds();
    __syncthreads();

    int buf = 0;
    for (int k0 = 0; k0 < K; k0 += 32, buf ^= 1) {
        const int k1 = k0 + 32;
        if (k1 < K) {                    // overlap next-tile DMA with compute
            cp16_g2l(gA0 + k1, &shA[buf ^ 1][lOff0]);
            cp16_g2l(gA1 + k1, &shA[buf ^ 1][lOff1]);
            cp16_g2l(gB0 + k1, &shB[buf ^ 1][lOff0]);
            cp16_g2l(gB1 + k1, &shB[buf ^ 1][lOff1]);
        }

        v16bf a[2], b[4];
#pragma unroll
        for (int mi = 0; mi < 2; ++mi) {
            const int r = (waveM * 32 + mi * 16 + l15) * LDS_STRIDE + h * 8;
            v8bf lo = *(const v8bf*)&shA[buf][r];
            v8bf hi = *(const v8bf*)&shA[buf][r + 16];
            a[mi] = __builtin_shufflevector(lo, hi,
                        0,1,2,3,4,5,6,7,8,9,10,11,12,13,14,15);
        }
#pragma unroll
        for (int ni = 0; ni < 4; ++ni) {
            const int r = (waveN * 64 + ni * 16 + l15) * LDS_STRIDE + h * 8;
            v8bf lo = *(const v8bf*)&shB[buf][r];
            v8bf hi = *(const v8bf*)&shB[buf][r + 16];
            b[ni] = __builtin_shufflevector(lo, hi,
                        0,1,2,3,4,5,6,7,8,9,10,11,12,13,14,15);
        }
#pragma unroll
        for (int mi = 0; mi < 2; ++mi)
#pragma unroll
            for (int ni = 0; ni < 4; ++ni)
                acc[mi][ni] = __builtin_amdgcn_wmma_f32_16x16x32_bf16(
                    false, a[mi], false, b[ni],
                    (short)0, acc[mi][ni], false, false);

        if (k1 < K) wait_async_lds();
        __syncthreads();
    }

    // C/D layout: VGPR r holds row (r + h*8) of the 16x16 tile, col = l15.
#pragma unroll
    for (int mi = 0; mi < 2; ++mi) {
#pragma unroll
        for (int r = 0; r < 8; ++r) {
            int row = blkM * 128 + waveM * 32 + mi * 16 + h * 8 + r;
#pragma unroll
            for (int ni = 0; ni < 4; ++ni) {
                int col = blkN * 128 + waveN * 64 + ni * 16 + l15;
                float v = acc[mi][ni][r];
                if (bias != nullptr && col < Nreal) v += bias[col];
                if (relu) v = fmaxf(v, 0.0f);
                if (Cf != nullptr && col < Nreal)
                    Cf[(size_t)row * Nreal + col] = v;
                if (Cb != nullptr)
                    Cb[(size_t)row * Npad + col] = cvt_bf16(v);
            }
        }
    }
}

extern "C" void kernel_launch(void* const* d_in, const int* in_sizes, int n_in,
                              void* d_out, int out_size, void* d_ws, size_t ws_size,
                              hipStream_t stream) {
    (void)in_sizes; (void)n_in; (void)out_size; (void)ws_size;

    const float* image = (const float*)d_in[0];  // [2048 x 2048]
    const float* wemb  = (const float*)d_in[1];  // [1000 x 300]
    const float* Wcls  = (const float*)d_in[2];  // [1000 x 2048]
    const float* bcls  = (const float*)d_in[3];  // [1000]
    const float* Wvis  = (const float*)d_in[4];  // [1024 x 2048]
    const float* bvis  = (const float*)d_in[5];  // [1024]
    const float* Wsem  = (const float*)d_in[6];  // [1024 x 300]
    const float* bsem  = (const float*)d_in[7];  // [1024]

    const int Bn = 2048, Cn = 1000, Cp = 1024, Vd = 2048,
              Hd = 1024, Sd = 300, Sp = 320;

    char* ws = (char*)d_ws;
    bf16_t* img_b  = (bf16_t*)ws;  ws += (size_t)Bn * Vd * 2;   // image bf16
    bf16_t* wcls_b = (bf16_t*)ws;  ws += (size_t)Cp * Vd * 2;   // W_cls padded
    bf16_t* wvis_b = (bf16_t*)ws;  ws += (size_t)Hd * Vd * 2;   // W_vis
    bf16_t* we_b   = (bf16_t*)ws;  ws += (size_t)Cp * Sp * 2;   // word emb padded
    bf16_t* wsem_b = (bf16_t*)ws;  ws += (size_t)Hd * Sp * 2;   // W_sem padded
    bf16_t* vemb_b = (bf16_t*)ws;  ws += (size_t)Bn * Hd * 2;   // visual_emb bf16
    bf16_t* semb_b = (bf16_t*)ws;  ws += (size_t)Cp * Hd * 2;   // semantic_emb bf16

    float* logits = (float*)d_out;                 // [2048 x 1000]
    float* scores = logits + (size_t)Bn * Cn;      // [2048 x 1000]

    auto cvt = [&](const float* s, int sr, int sc, bf16_t* d, int dr, int dc) {
        int total = dr * dc;
        cvt_pad_bf16<<<(total + 255) / 256, 256, 0, stream>>>(s, sr, sc, d, dr, dc);
    };
    cvt(image, Bn, Vd, img_b,  Bn, Vd);
    cvt(Wcls,  Cn, Vd, wcls_b, Cp, Vd);
    cvt(Wvis,  Hd, Vd, wvis_b, Hd, Vd);
    cvt(wemb,  Cn, Sd, we_b,   Cp, Sp);
    cvt(Wsem,  Hd, Sd, wsem_b, Hd, Sp);

    auto gemm = [&](const bf16_t* A, const bf16_t* B, const float* bias,
                    float* Cf, bf16_t* Cb, int M, int Npad, int Nreal, int K,
                    int relu) {
        int blocks = (M / 128) * (Npad / 128);
        gemm_bf16_wmma<<<blocks, 256, 0, stream>>>(A, B, bias, Cf, Cb,
                                                   M, Npad, Nreal, K, relu);
    };

    // logits = image @ W_cls^T + b_cls                        (fp32 out)
    gemm(img_b,  wcls_b, bcls,    logits, nullptr, Bn, Cp, Cn, Vd, 0);
    // visual_emb = relu(image @ W_vis^T + b_vis)              (bf16 out)
    gemm(img_b,  wvis_b, bvis,    nullptr, vemb_b, Bn, Hd, Hd, Vd, 1);
    // semantic_emb = relu(word_emb @ W_sem^T + b_sem)         (bf16 out)
    gemm(we_b,   wsem_b, bsem,    nullptr, semb_b, Cp, Hd, Hd, Sp, 1);
    // scores = visual_emb @ semantic_emb^T                    (fp32 out)
    gemm(vemb_b, semb_b, nullptr, scores,  nullptr, Bn, Cp, Cn, Hd, 0);
}
